// DeepAttMISL_27204322853675
// MI455X (gfx1250) — matrix operand order repaired and checked by reading, not compiled, not run on hardware
//
#include <hip/hip_runtime.h>
#include <hip/hip_bf16.h>

#define NUM_CLUSTERS 8
#define DIM_IN 1024
#define DIM_HID 256
#define NTOT 65536
#define ROWS_PER_WG 32

typedef __attribute__((ext_vector_type(16))) __bf16 v16bf;
typedef __attribute__((ext_vector_type(8)))  float  v8f;

union BfFrag {
    v16bf v;
    __bf16 h[16];
};

// ---------------------------------------------------------------------------
// k0: zero segment accumulators in workspace
// ---------------------------------------------------------------------------
__global__ void __launch_bounds__(256) k0_zero(float* p, int n) {
    for (int i = threadIdx.x; i < n; i += 256) p[i] = 0.0f;
}

// ---------------------------------------------------------------------------
// kw: one-time W1 f32 -> bf16 conversion (512 KB, lives in L2 thereafter)
// ---------------------------------------------------------------------------
__global__ void __launch_bounds__(256) kw_convert(const float* __restrict__ W1,
                                                  __bf16* __restrict__ w1bf) {
    const int i = blockIdx.x * 256 + threadIdx.x;
    if (i < DIM_HID * DIM_IN) w1bf[i] = (__bf16)W1[i];
}

// ---------------------------------------------------------------------------
// k1: fused  h = relu(x @ W1^T + b1)  + per-cluster segment sum/count.
// h is never materialized; only [8][256] sums + [8] counts leave the WG.
// ---------------------------------------------------------------------------
__global__ void __launch_bounds__(256)
k1_gemm_relu_segsum(const float*  __restrict__ x,
                    const int*    __restrict__ cid,
                    const __bf16* __restrict__ w1bf,   // [256][1024] bf16
                    const float*  __restrict__ b1,
                    float* __restrict__ seg_sum,       // [8][256]
                    float* __restrict__ seg_cnt) {     // [8]
    __shared__ float lds_seg[NUM_CLUSTERS][DIM_HID];   // 8 KB
    __shared__ float lds_cnt[NUM_CLUSTERS];
    __shared__ int   lds_cid[ROWS_PER_WG];

    const int tid  = threadIdx.x;
    const int wave = tid >> 5;
    const int lane = tid & 31;
    const int row0 = blockIdx.x * ROWS_PER_WG;

    for (int i = tid; i < NUM_CLUSTERS * DIM_HID; i += 256)
        (&lds_seg[0][0])[i] = 0.0f;
    if (tid < NUM_CLUSTERS)  lds_cnt[tid] = 0.0f;
    if (tid < ROWS_PER_WG)   lds_cid[tid] = cid[row0 + tid];
    __syncthreads();

    // tile assignment: 2 M-tiles x 16 N-tiles = 32 tiles / 8 waves = 4 each
    const int m_tile  = wave & 1;           // rows [m_tile*16, m_tile*16+16)
    const int n_base0 = (wave >> 1) * 64;   // 4 consecutive 16-col tiles
    const int lhalf   = lane >> 4;          // lane half (0|1)
    const int lmod    = lane & 15;

    const float* xrow = x + (size_t)(row0 + m_tile * 16 + lmod) * DIM_IN;

    v8f zero8 = {0.f, 0.f, 0.f, 0.f, 0.f, 0.f, 0.f, 0.f};
    v8f acc[4];
    acc[0] = zero8; acc[1] = zero8; acc[2] = zero8; acc[3] = zero8;

    for (int kb = 0; kb < DIM_IN; kb += 32) {
        // --- A fragment (16-bit A 16x32): element i of the
        // fragment = x[kb + lhalf*16 + i] -> two contiguous
        // 8-float runs.  Stream x non-temporally (read once, > L2).
        const v8f lo = __builtin_nontemporal_load(
            (const v8f*)(xrow + kb + lhalf * 16));
        const v8f hi = __builtin_nontemporal_load(
            (const v8f*)(xrow + kb + lhalf * 16 + 8));
        BfFrag a;
#pragma unroll
        for (int i = 0; i < 8; ++i) {
            a.h[i]     = (__bf16)lo[i];   // -> v_cvt bf16 (pairable)
            a.h[8 + i] = (__bf16)hi[i];
        }
        if (kb + 32 < DIM_IN)
            __builtin_prefetch(xrow + kb + 32, 0, 0);  // global_prefetch_b8

#pragma unroll
        for (int t = 0; t < 4; ++t) {
            // --- B fragment (16-bit B 32x16): lane = column n_base0+t*16+lmod,
            // K-range [kb + lhalf*16, +16) contiguous -> single 32B load of
            // pre-converted bf16 weights, zero conversion VALU.
            const __bf16* wcol =
                w1bf + (size_t)(n_base0 + t * 16 + lmod) * DIM_IN + kb + lhalf * 16;
            BfFrag b;
            b.v = *(const v16bf*)wcol;
            acc[t] = __builtin_amdgcn_wmma_f32_16x16x32_bf16(
                false, a.v, false, b.v, (short)0, acc[t], false, false);
        }
    }

    // --- epilogue: bias + ReLU + segment accumulate ---
    // C/D layout: VGPR r, lanes 0-15 -> M=r, lanes 16-31 -> M=8+r; N = L%16.
#pragma unroll
    for (int t = 0; t < 4; ++t) {
        const int j = n_base0 + t * 16 + lmod;
        const float bias = b1[j];
#pragma unroll
        for (int r = 0; r < 8; ++r) {
            const int mloc = m_tile * 16 + lhalf * 8 + r;
            float v = acc[t][r] + bias;
            v = v > 0.0f ? v : 0.0f;
            atomicAdd(&lds_seg[lds_cid[mloc]][j], v);   // ds_add_f32
        }
    }
    if (tid < ROWS_PER_WG) atomicAdd(&lds_cnt[lds_cid[tid]], 1.0f);
    __syncthreads();

    for (int i = tid; i < NUM_CLUSTERS * DIM_HID; i += 256)
        atomicAdd(&seg_sum[i], (&lds_seg[0][0])[i]);    // global_atomic_add_f32
    if (tid < NUM_CLUSTERS) atomicAdd(&seg_cnt[tid], lds_cnt[tid]);
}

// ---------------------------------------------------------------------------
// k2: tiny attention tail on the [8][256] cluster means (single workgroup)
// ---------------------------------------------------------------------------
__global__ void __launch_bounds__(256)
k2_finalize(const float* __restrict__ seg_sum, const float* __restrict__ seg_cnt,
            const float* __restrict__ Wf, const float* __restrict__ bfv,
            const float* __restrict__ Wa, const float* __restrict__ ba,
            const float* __restrict__ Wb, const float* __restrict__ bb,
            const float* __restrict__ Wc, const float* __restrict__ bc,
            float* __restrict__ out) {
    __shared__ float hc[NUM_CLUSTERS][DIM_HID];
    __shared__ float hp[NUM_CLUSTERS][DIM_HID];
    __shared__ float ag[NUM_CLUSTERS][DIM_HID];
    __shared__ float logits[NUM_CLUSTERS];
    __shared__ float wts[NUM_CLUSTERS];
    const int tid = threadIdx.x;  // = output column j

    for (int c = 0; c < NUM_CLUSTERS; ++c) {
        float cnt = seg_cnt[c];
        cnt = cnt > 1.0f ? cnt : 1.0f;
        hc[c][tid] = seg_sum[c * DIM_HID + tid] / cnt;
    }
    __syncthreads();

    for (int c = 0; c < NUM_CLUSTERS; ++c) {
        float s = bfv[tid];
        for (int d = 0; d < DIM_HID; ++d) s += Wf[tid * DIM_HID + d] * hc[c][d];
        hp[c][tid] = s > 0.0f ? s : 0.0f;
    }
    __syncthreads();

    for (int c = 0; c < NUM_CLUSTERS; ++c) {
        float sa = ba[tid], sg = bb[tid];
        for (int d = 0; d < DIM_HID; ++d) {
            const float h = hp[c][d];
            sa += Wa[tid * DIM_HID + d] * h;
            sg += Wb[tid * DIM_HID + d] * h;
        }
        ag[c][tid] = tanhf(sa) * (1.0f / (1.0f + __expf(-sg)));
    }
    __syncthreads();

    if (tid < NUM_CLUSTERS) {
        float s = bc[0];
        for (int d = 0; d < DIM_HID; ++d) s += ag[tid][d] * Wc[d];
        logits[tid] = s;
    }
    __syncthreads();
    if (tid == 0) {
        float m = logits[0];
        for (int c = 1; c < NUM_CLUSTERS; ++c) m = fmaxf(m, logits[c]);
        float z = 0.0f;
        for (int c = 0; c < NUM_CLUSTERS; ++c) { wts[c] = __expf(logits[c] - m); z += wts[c]; }
        for (int c = 0; c < NUM_CLUSTERS; ++c) wts[c] /= z;
    }
    __syncthreads();

    float o = 0.0f;
    for (int c = 0; c < NUM_CLUSTERS; ++c) o += wts[c] * hp[c][tid];
    out[tid] = o;
}

// ---------------------------------------------------------------------------
extern "C" void kernel_launch(void* const* d_in, const int* in_sizes, int n_in,
                              void* d_out, int out_size, void* d_ws, size_t ws_size,
                              hipStream_t stream) {
    const float* x   = (const float*)d_in[0];
    const int*   cid = (const int*)  d_in[1];
    const float* W1  = (const float*)d_in[2];
    const float* b1  = (const float*)d_in[3];
    const float* Wf  = (const float*)d_in[4];
    const float* bfv = (const float*)d_in[5];
    const float* Wa  = (const float*)d_in[6];
    const float* ba  = (const float*)d_in[7];
    const float* Wb  = (const float*)d_in[8];
    const float* bb  = (const float*)d_in[9];
    const float* Wc  = (const float*)d_in[10];
    const float* bc  = (const float*)d_in[11];

    // workspace layout
    float*  seg_sum = (float*)d_ws;                        // [8][256]
    float*  seg_cnt = seg_sum + NUM_CLUSTERS * DIM_HID;    // [8]
    __bf16* w1bf    = (__bf16*)((char*)d_ws + 8448);       // 256B-aligned, 512 KB

    k0_zero<<<1, 256, 0, stream>>>(seg_sum, NUM_CLUSTERS * DIM_HID + NUM_CLUSTERS);
    kw_convert<<<(DIM_HID * DIM_IN) / 256, 256, 0, stream>>>(W1, w1bf);

    k1_gemm_relu_segsum<<<NTOT / ROWS_PER_WG, 256, 0, stream>>>(
        x, cid, w1bf, b1, seg_sum, seg_cnt);

    k2_finalize<<<1, 256, 0, stream>>>(seg_sum, seg_cnt, Wf, bfv, Wa, ba,
                                       Wb, bb, Wc, bc, (float*)d_out);
}
